// SharedSparseMoE_81277961110151
// MI455X (gfx1250) — compile-verified, matrix operand order checked by reading, not run on hardware
//
#include <hip/hip_runtime.h>
#include <hip/hip_bf16.h>
#include <stdint.h>

#define D_DIM 1024
#define H_DIM 1408
#define E_NUM 8
#define T_TOK 4096
#define HP    1424   // padded LDS row stride for h tile (2848B, 16B-aligned)
#define XP    1040   // padded LDS row stride for x tile (2080B, 16B-aligned)

typedef __attribute__((ext_vector_type(16))) __bf16 v16bf;
typedef __attribute__((ext_vector_type(8)))  __bf16 v8bf;
typedef __attribute__((ext_vector_type(8)))  float  v8f;

// Fragment layout per ISA 7.12.2 (16-bit A 16x32 / symmetric B 32x16):
// lane = {idx%16, half=lane/16}; elements j=0..7 -> K = half*8 + j,
// elements j=8..15 -> K = 16 + half*8 + (j-8). base must point at (row, half*8).
__device__ __forceinline__ v16bf load_frag(const __bf16* base) {
  v8bf lo = *(const v8bf*)(base);
  v8bf hi = *(const v8bf*)(base + 16);
  v16bf r;
#pragma unroll
  for (int j = 0; j < 8; ++j) { r[j] = lo[j]; r[j + 8] = hi[j]; }
  return r;
}

// CDNA5 async global->LDS copy (16B per lane), tracked by ASYNCcnt.
// lds_off = LDS byte offset (low 32 bits of a generic pointer into __shared__),
// gptr    = 64-bit global address. GV addressing mode (SADDR=NULL -> "off").
__device__ __forceinline__ void async_copy_b128(uint32_t lds_off, const void* gptr) {
  asm volatile("global_load_async_to_lds_b128 %0, %1, off"
               :: "v"(lds_off), "v"(gptr)
               : "memory");
}
__device__ __forceinline__ void wait_asynccnt0() {
  asm volatile("s_wait_asynccnt 0x0" ::: "memory");
}

// ---------------------------------------------------------------------------
// Kernel 1: fp32 -> bf16 conversion; weights stored N-major (transposed) so
// WMMA B fragments are contiguous 16B loads.
// ---------------------------------------------------------------------------
__global__ __launch_bounds__(256)
void moe_convert_kernel(const float* __restrict__ x,
                        const float* __restrict__ w1, const float* __restrict__ w3,
                        const float* __restrict__ w2,
                        const float* __restrict__ sw1, const float* __restrict__ sw3,
                        const float* __restrict__ sw2,
                        __bf16* __restrict__ xbf,
                        __bf16* __restrict__ w1t, __bf16* __restrict__ w3t,
                        __bf16* __restrict__ w2t,
                        __bf16* __restrict__ sw1t, __bf16* __restrict__ sw3t,
                        __bf16* __restrict__ sw2t) {
  const unsigned NX = (unsigned)T_TOK * D_DIM;        // 4,194,304
  const unsigned NS = (unsigned)D_DIM * H_DIM;        // 1,441,792
  const unsigned NW = (unsigned)E_NUM * NS;           // 11,534,336
  const unsigned TOTAL = NX + 3u * NW + 3u * NS;
  const unsigned stride = gridDim.x * blockDim.x;
  for (unsigned i = blockIdx.x * blockDim.x + threadIdx.x; i < TOTAL; i += stride) {
    unsigned j = i;
    if (j < NX) { xbf[j] = (__bf16)x[j]; continue; }
    j -= NX;
    if (j < NW) {  // w1t[e][h][d] = w1[e][d][h]
      unsigned e = j / NS, rm = j % NS, h = rm / D_DIM, d = rm % D_DIM;
      w1t[j] = (__bf16)w1[(size_t)e * NS + (size_t)d * H_DIM + h];
      continue;
    }
    j -= NW;
    if (j < NW) {  // w3t[e][h][d] = w3[e][d][h]
      unsigned e = j / NS, rm = j % NS, h = rm / D_DIM, d = rm % D_DIM;
      w3t[j] = (__bf16)w3[(size_t)e * NS + (size_t)d * H_DIM + h];
      continue;
    }
    j -= NW;
    if (j < NW) {  // w2t[e][d][h] = w2[e][h][d]
      unsigned e = j / NS, rm = j % NS, d = rm / H_DIM, h = rm % H_DIM;
      w2t[j] = (__bf16)w2[(size_t)e * NS + (size_t)h * D_DIM + d];
      continue;
    }
    j -= NW;
    if (j < NS) {  // sw1t[h][d] = sw1[d][h]
      unsigned h = j / D_DIM, d = j % D_DIM;
      sw1t[j] = (__bf16)sw1[(size_t)d * H_DIM + h];
      continue;
    }
    j -= NS;
    if (j < NS) {  // sw3t[h][d] = sw3[d][h]
      unsigned h = j / D_DIM, d = j % D_DIM;
      sw3t[j] = (__bf16)sw3[(size_t)d * H_DIM + h];
      continue;
    }
    j -= NS;
    {              // sw2t[d][h] = sw2[h][d]
      unsigned d = j / H_DIM, h = j % H_DIM;
      sw2t[j] = (__bf16)sw2[(size_t)h * D_DIM + d];
    }
  }
}

// ---------------------------------------------------------------------------
// Kernel 2: gate. One wave per token: logits -> softmax -> top-2 (lowest-index
// tie break, matching lax.top_k) -> dense routing-weight row rw[t][0..7].
// ---------------------------------------------------------------------------
__global__ __launch_bounds__(256)
void moe_gate_kernel(const float* __restrict__ x, const float* __restrict__ gw,
                     const float* __restrict__ gb, float* __restrict__ rw) {
  const int lane = threadIdx.x & 31;
  const int wave = threadIdx.x >> 5;
  const int t = blockIdx.x * 8 + wave;
  if (t >= T_TOK) return;

  float acc[E_NUM];
#pragma unroll
  for (int e = 0; e < E_NUM; ++e) acc[e] = 0.0f;
  const float* xr = x + (size_t)t * D_DIM;
  for (int i = lane; i < D_DIM; i += 32) {
    float xv = xr[i];
#pragma unroll
    for (int e = 0; e < E_NUM; ++e) acc[e] += xv * gw[e * D_DIM + i];
  }
#pragma unroll
  for (int off = 16; off > 0; off >>= 1) {
#pragma unroll
    for (int e = 0; e < E_NUM; ++e) acc[e] += __shfl_xor(acc[e], off, 32);
  }
  if (lane == 0) {
    float m = acc[0];
#pragma unroll
    for (int e = 1; e < E_NUM; ++e) m = fmaxf(m, acc[e]);
    float sc[E_NUM], s = 0.0f;
#pragma unroll
    for (int e = 0; e < E_NUM; ++e) { sc[e] = __expf(acc[e] - m); s += sc[e]; }
    float inv = 1.0f / s;
    float bz[E_NUM];
#pragma unroll
    for (int e = 0; e < E_NUM; ++e) { sc[e] *= inv; bz[e] = sc[e] + gb[e]; }
    int i0 = 0;
#pragma unroll
    for (int e = 1; e < E_NUM; ++e) if (bz[e] > bz[i0]) i0 = e;
    int i1 = (i0 == 0) ? 1 : 0;
#pragma unroll
    for (int e = 0; e < E_NUM; ++e) if (e != i0 && bz[e] > bz[i1]) i1 = e;
    float row[E_NUM];
#pragma unroll
    for (int e = 0; e < E_NUM; ++e) row[e] = 0.0f;
    row[i0] = sc[i0];
    row[i1] = sc[i1];
    float* dst = rw + (size_t)t * E_NUM;
#pragma unroll
    for (int e = 0; e < E_NUM; ++e) dst[e] = row[e];
  }
}

// ---------------------------------------------------------------------------
// Kernel 3: fused expert compute. 1 WG = 16 tokens, 8 waves.
//  - x tile (16x1024 bf16, 32KB) staged ONCE into LDS via
//    global_load_async_to_lds_b128 (ASYNCcnt); reused ~99x by GEMM1 A-frags.
//  - For shared expert + every routed expert active in the tile:
//      GEMM1: h[16,1408] = silu((x W1 + b1) * (x W3 + b3))  (bf16 WMMA, f32 acc)
//      GEMM2: out += rw * (h W2 + b2)                        (A from LDS)
// ---------------------------------------------------------------------------
__global__ __launch_bounds__(256)
void moe_expert_kernel(const __bf16* __restrict__ xbf,
                       const __bf16* __restrict__ w1t, const __bf16* __restrict__ w3t,
                       const __bf16* __restrict__ w2t,
                       const __bf16* __restrict__ sw1t, const __bf16* __restrict__ sw3t,
                       const __bf16* __restrict__ sw2t,
                       const float* __restrict__ b1, const float* __restrict__ b3,
                       const float* __restrict__ b2,
                       const float* __restrict__ sb1, const float* __restrict__ sb3,
                       const float* __restrict__ sb2,
                       const float* __restrict__ rw, float* __restrict__ out) {
  __shared__ __attribute__((aligned(16))) __bf16 hlds[16 * HP];  // 45.6 KB
  __shared__ __attribute__((aligned(16))) __bf16 xlds[16 * XP];  // 33.3 KB
  __shared__ float rwtile[16][E_NUM];
  __shared__ int active[E_NUM + 1];

  const int tid  = threadIdx.x;
  const int lane = tid & 31;
  const int wave = tid >> 5;
  const int r    = lane & 15;   // M (A,C/D) or N (B,C/D) index within tile
  const int half = lane >> 4;   // K-half selector
  const int t0   = blockIdx.x * 16;

  // ---- async-stage x tile: 16 rows x 1024 bf16 = 2048 x 16B chunks ----
  {
#pragma unroll
    for (int it = 0; it < 8; ++it) {
      const int c    = tid + it * 256;       // chunk id 0..2047
      const int row  = c >> 7;               // 0..15
      const int col8 = (c & 127) * 8;        // bf16 element offset in row
      const uint32_t ldst = (uint32_t)(uintptr_t)(xlds + row * XP + col8);
      const void* gsrc = (const void*)(xbf + (size_t)(t0 + row) * D_DIM + col8);
      async_copy_b128(ldst, gsrc);
    }
  }

  if (tid <= E_NUM) active[tid] = (tid == E_NUM) ? 1 : 0;
  __syncthreads();
  if (tid < 16 * E_NUM) {
    int i = tid >> 3, j = tid & 7;
    float v = rw[(size_t)(t0 + i) * E_NUM + j];
    rwtile[i][j] = v;
    if (v != 0.0f) atomicOr(&active[j], 1);
  }
  wait_asynccnt0();   // x tile resident in LDS
  __syncthreads();

  v8f outacc[8];
#pragma unroll
  for (int ct = 0; ct < 8; ++ct) outacc[ct] = {};

  const size_t NS = (size_t)D_DIM * H_DIM;

  for (int e = 0; e <= E_NUM; ++e) {
    if (!active[e]) continue;  // uniform across the workgroup (shared mem)
    const __bf16* B1  = (e < E_NUM) ? (w1t + (size_t)e * NS) : sw1t;
    const __bf16* B3  = (e < E_NUM) ? (w3t + (size_t)e * NS) : sw3t;
    const __bf16* B2  = (e < E_NUM) ? (w2t + (size_t)e * NS) : sw2t;
    const float*  bb1 = (e < E_NUM) ? (b1 + e * H_DIM) : sb1;
    const float*  bb3 = (e < E_NUM) ? (b3 + e * H_DIM) : sb3;
    const float*  bb2 = (e < E_NUM) ? (b2 + e * D_DIM) : sb2;

    // ---- GEMM1: 88 column tiles over 8 waves, K = D_DIM, A from LDS ----
    const __bf16* aBase = xlds + (size_t)r * XP + half * 8;
    for (int ct = wave; ct < H_DIM / 16; ct += 8) {
      const int n0 = ct * 16;
      const __bf16* b1Base = B1 + (size_t)(n0 + r) * D_DIM + half * 8;
      const __bf16* b3Base = B3 + (size_t)(n0 + r) * D_DIM + half * 8;
      v8f acc1 = {}, acc3 = {};
      for (int k0 = 0; k0 < D_DIM; k0 += 32) {
        __builtin_prefetch((const void*)(b1Base + k0 + 128), 0, 0);
        __builtin_prefetch((const void*)(b3Base + k0 + 128), 0, 0);
        v16bf a  = load_frag(aBase + k0);    // ds_load_b128 x2
        v16bf f1 = load_frag(b1Base + k0);   // global_load_b128 x2
        v16bf f3 = load_frag(b3Base + k0);
        acc1 = __builtin_amdgcn_wmma_f32_16x16x32_bf16(false, a, false, f1,
                                                       (short)0, acc1, false, false);
        acc3 = __builtin_amdgcn_wmma_f32_16x16x32_bf16(false, a, false, f3,
                                                       (short)0, acc3, false, false);
      }
      const float bias1 = bb1[n0 + r];
      const float bias3 = bb3[n0 + r];
#pragma unroll
      for (int i = 0; i < 8; ++i) {
        float p = (acc1[i] + bias1) * (acc3[i] + bias3);
        float hv = p / (1.0f + __expf(-p));  // silu(p) -- silu on the PRODUCT
        hlds[(size_t)(i + 8 * half) * HP + (n0 + r)] = (__bf16)hv;
      }
    }
    __syncthreads();

    // ---- GEMM2: 64 column tiles (8 per wave), K = H_DIM, A from LDS ----
    const __bf16* a2Base = hlds + (size_t)r * HP + half * 8;
#pragma unroll
    for (int ct = 0; ct < 8; ++ct) {
      const int n0 = (wave * 8 + ct) * 16;
      const __bf16* b2Base = B2 + (size_t)(n0 + r) * H_DIM + half * 8;
      v8f acc = {};
      for (int k0 = 0; k0 < H_DIM; k0 += 32) {
        __builtin_prefetch((const void*)(b2Base + k0 + 128), 0, 0);
        v16bf a = load_frag(a2Base + k0);   // ds_load_b128 x2
        v16bf b = load_frag(b2Base + k0);   // global_load_b128 x2
        acc = __builtin_amdgcn_wmma_f32_16x16x32_bf16(false, a, false, b,
                                                      (short)0, acc, false, false);
      }
      const float bias2 = bb2[n0 + r];
#pragma unroll
      for (int i = 0; i < 8; ++i) {
        const int row = i + 8 * half;
        const float wgt = (e < E_NUM) ? rwtile[row][e] : 1.0f;
        outacc[ct][i] += wgt * (acc[i] + bias2);
      }
    }
    __syncthreads();  // hlds reused by next expert
  }

  // ---- store 16x1024 fp32 tile, each element written exactly once ----
#pragma unroll
  for (int ct = 0; ct < 8; ++ct) {
    const int n0 = (wave * 8 + ct) * 16;
#pragma unroll
    for (int i = 0; i < 8; ++i) {
      const int row = i + 8 * half;
      out[(size_t)(t0 + row) * D_DIM + (n0 + r)] = outacc[ct][i];
    }
  }
}

// ---------------------------------------------------------------------------
extern "C" void kernel_launch(void* const* d_in, const int* in_sizes, int n_in,
                              void* d_out, int out_size, void* d_ws, size_t ws_size,
                              hipStream_t stream) {
  const float* x   = (const float*)d_in[0];
  const float* gw  = (const float*)d_in[1];
  const float* gb  = (const float*)d_in[2];
  const float* w1  = (const float*)d_in[3];
  const float* b1  = (const float*)d_in[4];
  const float* w3  = (const float*)d_in[5];
  const float* b3  = (const float*)d_in[6];
  const float* w2  = (const float*)d_in[7];
  const float* b2  = (const float*)d_in[8];
  const float* sw1 = (const float*)d_in[9];
  const float* sb1 = (const float*)d_in[10];
  const float* sw3 = (const float*)d_in[11];
  const float* sb3 = (const float*)d_in[12];
  const float* sw2 = (const float*)d_in[13];
  const float* sb2 = (const float*)d_in[14];
  float* out = (float*)d_out;

  const size_t NX = (size_t)T_TOK * D_DIM;
  const size_t NS = (size_t)D_DIM * H_DIM;
  const size_t NW = (size_t)E_NUM * NS;

  uintptr_t p = (uintptr_t)d_ws;
  auto take = [&p](size_t bytes) -> uintptr_t {
    uintptr_t r = p;
    p += (bytes + 255) & ~(size_t)255;
    return r;
  };
  __bf16* xbf  = (__bf16*)take(NX * 2);
  __bf16* w1t  = (__bf16*)take(NW * 2);
  __bf16* w3t  = (__bf16*)take(NW * 2);
  __bf16* w2t  = (__bf16*)take(NW * 2);
  __bf16* sw1t = (__bf16*)take(NS * 2);
  __bf16* sw3t = (__bf16*)take(NS * 2);
  __bf16* sw2t = (__bf16*)take(NS * 2);
  float*  rw   = (float*)take((size_t)T_TOK * E_NUM * 4);

  moe_convert_kernel<<<dim3(8192), dim3(256), 0, stream>>>(
      x, w1, w3, w2, sw1, sw3, sw2, xbf, w1t, w3t, w2t, sw1t, sw3t, sw2t);
  moe_gate_kernel<<<dim3(T_TOK / 8), dim3(256), 0, stream>>>(x, gw, gb, rw);
  moe_expert_kernel<<<dim3(T_TOK / 16), dim3(256), 0, stream>>>(
      xbf, w1t, w3t, w2t, sw1t, sw3t, sw2t,
      b1, b3, b2, sb1, sb3, sb2, rw, out);
}
